// GATLayer_17489106829984
// MI455X (gfx1250) — matrix operand order repaired and checked by reading, not compiled, not run on hardware
//
#include <hip/hip_runtime.h>
#include <cstdint>

#define C_IN 256
#define C_OUTC 256
#define NUM_HEADS 4
#define C_HEAD 64
#define ALPHA 0.2f
#define NEG_FILL (-9.0e15f)

#define USE_ASYNC_LDS 1

typedef __attribute__((ext_vector_type(16))) _Float16 v16h;
typedef __attribute__((ext_vector_type(8)))  _Float16 v8h;
typedef __attribute__((ext_vector_type(8)))  float    v8f;
typedef __attribute__((ext_vector_type(4)))  float    v4f;
typedef __attribute__((ext_vector_type(4)))  int      v4i;

__device__ __forceinline__ float leaky(float v) { return v >= 0.f ? v : ALPHA * v; }

// ---------------------------------------------------------------------------
// Kernel 0: WT16[col][k] = (f16) W[k][col]   (256x256, tiny)
// ---------------------------------------------------------------------------
__global__ void prep_kernel(const float* __restrict__ W, _Float16* __restrict__ WT16) {
  const int col = blockIdx.x * blockDim.x + threadIdx.x;
  if (col >= C_OUTC) return;
  for (int k = 0; k < C_IN; ++k)
    WT16[col * C_IN + k] = (_Float16)W[k * C_OUTC + col];
}

// ---------------------------------------------------------------------------
// Kernel 1: feats = X@W + b via WMMA.
//   - writes featsT16[col][j]  (f16, transposed, for phase-2 B operands)
//   - writes srcW[h][j], dstW[h][j] (attention scalars)
// grid = N/16 blocks x 128 threads (wave h owns head h's 64 output columns)
// ---------------------------------------------------------------------------
__global__ void feats_kernel(const float* __restrict__ X,
                             const _Float16* __restrict__ WT16,
                             const float* __restrict__ bvec,
                             const float* __restrict__ avec,
                             _Float16* __restrict__ featsT16,
                             float* __restrict__ srcW,
                             float* __restrict__ dstW, int N) {
  const int lane = threadIdx.x & 31;
  const int h    = threadIdx.x >> 5;     // wave == head
  const int i0   = blockIdx.x * 16;
  const int n    = lane & 15;            // A row / B col / D col
  const int hi   = lane >> 4;
  const int kbA  = hi * 8;               // A-layout K base (runs kbA..+7, kbA+16..+23)
  const int kbB  = hi * 8;               // B-layout K base (runs kbB..+7, kbB+16..+23)
  const int roff = hi * 8;               // D-layout row offset

  v8f acc[4];
  float asrc[4], adst[4];
#pragma unroll
  for (int ct = 0; ct < 4; ++ct) {
    const int col  = h * C_HEAD + ct * 16 + n;
    const float bv = bvec[col];
#pragma unroll
    for (int r = 0; r < 8; ++r) acc[ct][r] = bv;      // bias folded into C init
    asrc[ct] = avec[h * (2 * C_HEAD) + ct * 16 + n];
    adst[ct] = avec[h * (2 * C_HEAD) + C_HEAD + ct * 16 + n];
  }

  const float* xrow = X + (size_t)(i0 + n) * C_IN;
  for (int k0 = 0; k0 < C_IN; k0 += 32) {
    v4f x0 = *(const v4f*)(xrow + k0 + kbA);
    v4f x1 = *(const v4f*)(xrow + k0 + kbA + 4);
    v4f x2 = *(const v4f*)(xrow + k0 + kbA + 16);
    v4f x3 = *(const v4f*)(xrow + k0 + kbA + 20);
    v16h amat;
#pragma unroll
    for (int e = 0; e < 4; ++e) {
      amat[e]      = (_Float16)x0[e];
      amat[4 + e]  = (_Float16)x1[e];
      amat[8 + e]  = (_Float16)x2[e];
      amat[12 + e] = (_Float16)x3[e];
    }
#pragma unroll
    for (int ct = 0; ct < 4; ++ct) {
      const _Float16* bp = WT16 + (size_t)(h * C_HEAD + ct * 16 + n) * C_IN + k0 + kbB;
      const v8h b0 = *(const v8h*)(bp);
      const v8h b1 = *(const v8h*)(bp + 16);
      const v16h bmat = __builtin_shufflevector(b0, b1, 0,1,2,3,4,5,6,7,8,9,10,11,12,13,14,15);
      acc[ct] = __builtin_amdgcn_wmma_f32_16x16x32_f16(false, amat, false, bmat,
                                                       (short)0, acc[ct], false, false);
    }
  }

  // transposed f16 feats: 8 consecutive rows per lane -> one 16B store per ctile
#pragma unroll
  for (int ct = 0; ct < 4; ++ct) {
    v8h p;
#pragma unroll
    for (int r = 0; r < 8; ++r) p[r] = (_Float16)acc[ct][r];
    *(v8h*)(featsT16 + (size_t)(h * C_HEAD + ct * 16 + n) * N + i0 + roff) = p;
  }

  // src/dst scalars: reduce D-layout columns across the 16-lane half
  float sp[8], dp[8];
#pragma unroll
  for (int r = 0; r < 8; ++r) {
    float s = 0.f, d = 0.f;
#pragma unroll
    for (int ct = 0; ct < 4; ++ct) { s += acc[ct][r] * asrc[ct]; d += acc[ct][r] * adst[ct]; }
    sp[r] = s; dp[r] = d;
  }
#pragma unroll
  for (int m = 1; m <= 8; m <<= 1) {
#pragma unroll
    for (int r = 0; r < 8; ++r) {
      sp[r] += __shfl_xor(sp[r], m, 32);
      dp[r] += __shfl_xor(dp[r], m, 32);
    }
  }
  if (n == 0) {
#pragma unroll
    for (int r = 0; r < 8; ++r) {
      srcW[(size_t)h * N + i0 + roff + r] = sp[r];
      dstW[(size_t)h * N + i0 + roff + r] = dp[r];
    }
  }
}

// ---------------------------------------------------------------------------
// Kernel 2: attention + aggregation. One wave per (16-row block, head).
// grid = (N/16, NUM_HEADS), block = 32.
// Pass1: row max. Pass2: P~ = exp(logit-m) built in WMMA A layout,
// out = (P~ @ featsT16^T) / rowsum.
// ---------------------------------------------------------------------------
__global__ void attn_kernel(const int* __restrict__ adj,
                            const _Float16* __restrict__ featsT16,
                            const float* __restrict__ srcW,
                            const float* __restrict__ dstW,
                            float* __restrict__ out, int N) {
  __shared__ float ldsDst[4096];
  __shared__ float ldsM[16];
  __shared__ float ldsL[16];

  const int lane = threadIdx.x & 31;
  const int h    = blockIdx.y;
  const int i0   = blockIdx.x * 16;
  const int n    = lane & 15;
  const int hi   = lane >> 4;
  const int kbA  = hi * 8;
  const int jbB  = hi * 8;
  const int roff = hi * 8;

  // ---- stage dst[h][*] into LDS (CDNA5 async global->LDS path) ----
  {
    const float* gsrc = dstW + (size_t)h * N;
    for (int it = 0; it < N / 128; ++it) {
      const int t = it * 128 + lane * 4;
#if USE_ASYNC_LDS
      uint32_t laddr = (uint32_t)(uintptr_t)(&ldsDst[t]);
      uint32_t goff  = (uint32_t)(t * 4);
      asm volatile("global_load_async_to_lds_b128 %0, %1, %2"
                   :: "v"(laddr), "v"(goff), "s"(gsrc) : "memory");
#else
      *(v4f*)(&ldsDst[t]) = *(const v4f*)(gsrc + t);
#endif
    }
#if USE_ASYNC_LDS
    asm volatile("s_wait_asynccnt 0" ::: "memory");
#endif
  }
  __syncthreads();

  // ---- pass 1: per-row max of masked leaky logits ----
  for (int r = 0; r < 16; ++r) {
    const float srcv = srcW[(size_t)h * N + i0 + r];
    const int* arow  = adj + (size_t)(i0 + r) * N;
    float m = NEG_FILL;
    for (int it = 0; it < N / 128; ++it) {
      const int j = it * 128 + lane * 4;
      v4i av = *(const v4i*)(arow + j);
      v4f dv = *(const v4f*)(&ldsDst[j]);
#pragma unroll
      for (int e = 0; e < 4; ++e) {
        float v = leaky(srcv + dv[e]);
        v = (av[e] == 1) ? v : NEG_FILL;
        m = fmaxf(m, v);
      }
    }
#pragma unroll
    for (int mm = 1; mm <= 16; mm <<= 1) m = fmaxf(m, __shfl_xor(m, mm, 32));
    if (lane == 0) ldsM[r] = m;
  }
  __syncthreads();

  // ---- pass 2: WMMA accumulation of exp(logit-m) @ feats ----
  const float srcv = srcW[(size_t)h * N + i0 + n];
  const float mrow = ldsM[n];
  const int* arow  = adj + (size_t)(i0 + n) * N;
  v8f acc[4];
#pragma unroll
  for (int ct = 0; ct < 4; ++ct)
#pragma unroll
    for (int r = 0; r < 8; ++r) acc[ct][r] = 0.f;
  float lsum = 0.f;

  for (int j0 = 0; j0 < N; j0 += 32) {
    __builtin_prefetch(arow + j0 + 512, 0, 0);
    v4i a0 = *(const v4i*)(arow + j0 + kbA);
    v4i a1 = *(const v4i*)(arow + j0 + kbA + 4);
    v4i a2 = *(const v4i*)(arow + j0 + kbA + 16);
    v4i a3 = *(const v4i*)(arow + j0 + kbA + 20);
    v4f d0 = *(const v4f*)(&ldsDst[j0 + kbA]);
    v4f d1 = *(const v4f*)(&ldsDst[j0 + kbA + 4]);
    v4f d2 = *(const v4f*)(&ldsDst[j0 + kbA + 16]);
    v4f d3 = *(const v4f*)(&ldsDst[j0 + kbA + 20]);
    v16h amat;
#pragma unroll
    for (int e = 0; e < 4; ++e) {
      float v0 = leaky(srcv + d0[e]); v0 = (a0[e] == 1) ? v0 : NEG_FILL;
      float p0 = __expf(v0 - mrow); lsum += p0; amat[e] = (_Float16)p0;
      float v1 = leaky(srcv + d1[e]); v1 = (a1[e] == 1) ? v1 : NEG_FILL;
      float p1 = __expf(v1 - mrow); lsum += p1; amat[4 + e] = (_Float16)p1;
      float v2 = leaky(srcv + d2[e]); v2 = (a2[e] == 1) ? v2 : NEG_FILL;
      float p2 = __expf(v2 - mrow); lsum += p2; amat[8 + e] = (_Float16)p2;
      float v3 = leaky(srcv + d3[e]); v3 = (a3[e] == 1) ? v3 : NEG_FILL;
      float p3 = __expf(v3 - mrow); lsum += p3; amat[12 + e] = (_Float16)p3;
    }
#pragma unroll
    for (int ct = 0; ct < 4; ++ct) {
      const _Float16* bp = featsT16 + (size_t)(h * C_HEAD + ct * 16 + n) * N + j0 + jbB;
      const v8h b0 = *(const v8h*)(bp);
      const v8h b1 = *(const v8h*)(bp + 16);
      const v16h bmat = __builtin_shufflevector(b0, b1, 0,1,2,3,4,5,6,7,8,9,10,11,12,13,14,15);
      acc[ct] = __builtin_amdgcn_wmma_f32_16x16x32_f16(false, amat, false, bmat,
                                                       (short)0, acc[ct], false, false);
    }
  }

  // lanes L and L+16 cover complementary K sets of row n -> combine
  lsum += __shfl_xor(lsum, 16, 32);
  if (lane < 16) ldsL[lane] = lsum;
  __syncthreads();

#pragma unroll
  for (int ct = 0; ct < 4; ++ct) {
#pragma unroll
    for (int r = 0; r < 8; ++r) {
      const int row    = roff + r;
      const float invl = 1.0f / ldsL[row];
      out[(size_t)(i0 + row) * C_OUTC + h * C_HEAD + ct * 16 + n] = acc[ct][r] * invl;
    }
  }
}

// ---------------------------------------------------------------------------
extern "C" void kernel_launch(void* const* d_in, const int* in_sizes, int n_in,
                              void* d_out, int out_size, void* d_ws, size_t ws_size,
                              hipStream_t stream) {
  (void)n_in; (void)out_size; (void)ws_size;
  const float* X   = (const float*)d_in[0];
  const int*   adj = (const int*)d_in[1];
  const float* W   = (const float*)d_in[2];
  const float* b   = (const float*)d_in[3];
  const float* a   = (const float*)d_in[4];
  float* out = (float*)d_out;
  const int N = in_sizes[0] / C_IN;   // 4096

  char* ws = (char*)d_ws;
  _Float16* featsT16 = (_Float16*)ws;                                  // 256*N f16
  float* srcW = (float*)(ws + (size_t)C_OUTC * N * sizeof(_Float16));  // H*N f32
  float* dstW = srcW + (size_t)NUM_HEADS * N;                          // H*N f32
  _Float16* WT16 = (_Float16*)(dstW + (size_t)NUM_HEADS * N);          // 256*256 f16

  prep_kernel<<<1, 256, 0, stream>>>(W, WT16);
  feats_kernel<<<N / 16, 128, 0, stream>>>(X, WT16, b, a, featsT16, srcW, dstW, N);
  attn_kernel<<<dim3(N / 16, NUM_HEADS), 32, 0, stream>>>(adj, featsT16, srcW, dstW, out, N);
}
